// RNN_26895085207932
// MI455X (gfx1250) — matrix operand order repaired and checked by reading, not compiled, not run on hardware
//
#include <hip/hip_runtime.h>
#include <hip/hip_bf16.h>

// Elman RNN, SEQ=2048, B=4096, IN=1, H=32.
// One wave (32 lanes) owns 16 batch rows for the whole sequence.
// Recurrence computed transposed:  D(h_out x batch) = W_hh x H^T  via
// v_wmma_f32_16x16x32_bf16 (A = W_hh rows, loop-invariant in VGPRs; B = H^T,
// rebuilt each step as packed-bf16 dwords with ds_swizzle SWAPX16 + cndmask).
// x load is software-pipelined one step ahead; x prefetched 8 steps ahead.

#define SEQ_N  2048
#define BATCH  4096
#define HDIM   32

typedef __attribute__((ext_vector_type(16))) __bf16 v16bf;
typedef __attribute__((ext_vector_type(2)))  __bf16 bf16x2;
typedef __attribute__((ext_vector_type(8)))  float  v8f;
typedef __attribute__((ext_vector_type(8)))  int    v8i;

static __device__ __forceinline__ int swz16i(int i) {
  // ds_swizzle SWAPX16: group-of-32, xor_mask=0x10, and_mask=0x1f
  return __builtin_amdgcn_ds_swizzle(i, 0x401F);
}

static __device__ __forceinline__ float swz16f(float v) {
  return __builtin_bit_cast(float, swz16i(__builtin_bit_cast(int, v)));
}

static __device__ __forceinline__ int pack_bf16(float a, float b) {
  bf16x2 p;
  p.x = (__bf16)a;
  p.y = (__bf16)b;   // -> v_cvt_pk_bf16_f32
  return __builtin_bit_cast(int, p);
}

static __device__ __forceinline__ float tanh_fast(float x) {
  // tanh(x) = 1 - 2/(exp(2x)+1); v_exp_f32 (2/ln2 scale) + v_rcp_f32
  float e = __builtin_amdgcn_exp2f(x * 2.885390081777927f);
  return 1.0f - 2.0f * __builtin_amdgcn_rcpf(e + 1.0f);
}

__global__ __launch_bounds__(32)
void rnn_elman_wmma(const float* __restrict__ x,
                    const float* __restrict__ hidden,
                    const float* __restrict__ W_ih,
                    const float* __restrict__ b_ih,
                    const float* __restrict__ W_hh,
                    const float* __restrict__ b_hh,
                    const float* __restrict__ W_fc,
                    const float* __restrict__ b_fc,
                    float* __restrict__ out) {
  const int lane = threadIdx.x & 31;
  const int g    = lane >> 4;     // lane half
  const int n    = lane & 15;     // batch column within tile / matrix row M
  const int b0   = blockIdx.x << 4;
  const bool hi  = (g != 0);

  // ---- A operands: W_hh rows, bf16 16x32 A-layout (loop invariant) ----
  // A operand, contiguous k order used here (B matches): lane M = lane%16; VGPR v holds K pair
  // K = 16*(lane/16) + 2*v, packed [15:0],[31:16].
  v16bf a0, a1;
#pragma unroll
  for (int v = 0; v < 8; ++v) {
    int K  = (g << 4) + (v << 1);
    a0[2*v]   = (__bf16)W_hh[n * HDIM + K];
    a0[2*v+1] = (__bf16)W_hh[n * HDIM + K + 1];
    a1[2*v]   = (__bf16)W_hh[(n + 16) * HDIM + K];
    a1[2*v+1] = (__bf16)W_hh[(n + 16) * HDIM + K + 1];
  }

  // ---- per-lane constants for C/D layout position h = r + 8g + 16t ----
  float wih_c[16], xb_c[16], wfc_c[16];
#pragma unroll
  for (int t = 0; t < 2; ++t)
#pragma unroll
    for (int r = 0; r < 8; ++r) {
      int h = r + 8 * g + 16 * t;
      wih_c[8*t + r] = W_ih[h];
      xb_c [8*t + r] = b_ih[h] + b_hh[h];
      wfc_c[8*t + r] = W_fc[h];
    }
  const float bfc = b_fc[0];

  // ---- initial B operand from `hidden`: B[k, n] = h0[b0+n, k] ----
  // B operand, same contiguous k order as A: dword r holds K = 16g + 2r (+parity), col N = n.
  v16bf hb;
#pragma unroll
  for (int e = 0; e < 16; ++e) {
    int K = (g << 4) + ((e >> 1) << 1) + (e & 1);
    hb[e] = (__bf16)hidden[(size_t)(b0 + n) * HDIM + K];
  }
  v8i bmi = __builtin_bit_cast(v8i, hb);

  // software-pipelined x load (one step ahead of use)
  int   ofs = 0;
  float xv  = x[b0 + n];

  for (int s = 0; s < SEQ_N; ++s) {
    // prefetch 8 steps ahead (clamped; speculative -> safe)
    int sp = (s + 8 < SEQ_N) ? (s + 8) : (SEQ_N - 1);
    __builtin_prefetch(&x[(size_t)sp * BATCH + b0 + n], 0, 1);  // global_prefetch_b8

    // C = x * W_ih + (b_ih + b_hh), in C/D layout
    v8f c0, c1;
#pragma unroll
    for (int r = 0; r < 8; ++r) {
      c0[r] = __builtin_fmaf(xv, wih_c[r],     xb_c[r]);
      c1[r] = __builtin_fmaf(xv, wih_c[8 + r], xb_c[8 + r]);
    }

    // D = W_hh x H^T + C  (two 16x16 output tiles: h_out 0..15 and 16..31)
    v16bf bm = __builtin_bit_cast(v16bf, bmi);
    v8f d0 = __builtin_amdgcn_wmma_f32_16x16x32_bf16(false, a0, false, bm,
                                                     (short)0, c0, false, false);
    v8f d1 = __builtin_amdgcn_wmma_f32_16x16x32_bf16(false, a1, false, bm,
                                                     (short)0, c1, false, false);

    // issue next step's x load now; the wait is absorbed by tanh/relayout below
    int nofs = (s + 1 < SEQ_N) ? (ofs + BATCH) : ofs;
    float xv_next = x[nofs + b0 + n];

    // h = tanh(D), kept f32
    float h0[8], h1[8];
#pragma unroll
    for (int r = 0; r < 8; ++r) { h0[r] = tanh_fast(d0[r]); h1[r] = tanh_fast(d1[r]); }

    // trailing Linear: out[s,b] = sum_h h[h,b] * W_fc[h] + b_fc
    // lane covers h = r+8g+16t; pair (lane, lane^16) covers all 32 h.
    float acc = 0.0f;
#pragma unroll
    for (int r = 0; r < 8; ++r) {
      acc = __builtin_fmaf(h0[r], wfc_c[r],     acc);
      acc = __builtin_fmaf(h1[r], wfc_c[8 + r], acc);
    }
    float full = acc + swz16f(acc) + bfc;
    if (lane < 16) out[ofs + b0 + n] = full;

    // rebuild B operand (H^T) for next step, on packed-bf16 dwords:
    //   dword r (r<4): g=0 wants pack(H[2r],H[2r+1])     = p0[r] here;
    //                  g=1 wants pack(H[16+2r],...)      = swz(p1[r]).
    //   dword r+4:     g=0 wants pack(H[8+2r],...)       = swz(p0[r]);
    //                  g=1 wants pack(H[24+2r],...)      = p1[r] here.
#pragma unroll
    for (int r = 0; r < 4; ++r) {
      int p0 = pack_bf16(h0[2*r], h0[2*r+1]);
      int p1 = pack_bf16(h1[2*r], h1[2*r+1]);
      int q0 = swz16i(p0);
      int q1 = swz16i(p1);
      bmi[r]     = hi ? q1 : p0;
      bmi[r + 4] = hi ? p1 : q0;
    }

    xv  = xv_next;
    ofs = nofs;
  }
}

extern "C" void kernel_launch(void* const* d_in, const int* in_sizes, int n_in,
                              void* d_out, int out_size, void* d_ws, size_t ws_size,
                              hipStream_t stream) {
  (void)in_sizes; (void)n_in; (void)out_size; (void)d_ws; (void)ws_size;
  const float* x      = (const float*)d_in[0];
  const float* hidden = (const float*)d_in[1];
  const float* W_ih   = (const float*)d_in[2];
  const float* b_ih   = (const float*)d_in[3];
  const float* W_hh   = (const float*)d_in[4];
  const float* b_hh   = (const float*)d_in[5];
  const float* W_fc   = (const float*)d_in[6];
  const float* b_fc   = (const float*)d_in[7];
  float* out = (float*)d_out;

  // one wave per 16-batch tile -> 256 independent waves across the chip
  rnn_elman_wmma<<<dim3(BATCH / 16), dim3(32), 0, stream>>>(
      x, hidden, W_ih, b_ih, W_hh, b_hh, W_fc, b_fc, out);
}
